// JKNet_5282809775006
// MI455X (gfx1250) — matrix operand order repaired and checked by reading, not compiled, not run on hardware
//
#include <hip/hip_runtime.h>
#include <hip/hip_bf16.h>

// ---------- types ----------
typedef __attribute__((ext_vector_type(16))) _Float16 v16h;
typedef __attribute__((ext_vector_type(8)))  _Float16 v8h;
typedef __attribute__((ext_vector_type(8)))  float    v8f;

#define HIDC 256
#define OUTC 47
#define BN_SCALE 0.9999950000374996f  /* rsqrt(1 + 1e-5) */

__device__ __forceinline__ int imind(int a, int b) { return a < b ? a : b; }

// ---------- WMMA operand loaders ----------
// A (MxK, f16): lane l holds row m; half-K pattern per ISA:
//   lanes 0-15:  K = k0 + {0..7, 16..23}
//   lanes 16-31: K = k0 + {8..15, 24..31}
__device__ __forceinline__ v16h load_a_f16(const _Float16* __restrict__ base, int ld,
                                           int m, int k0, int lane) {
  const _Float16* p = base + (size_t)m * ld + k0 + ((lane >> 4) << 3);
  union { v16h v; v8h h[2]; } u;
  u.h[0] = *(const v8h*)(p);
  u.h[1] = *(const v8h*)(p + 16);
  return u.v;
}

__device__ __forceinline__ v16h load_a_f32s(const float* __restrict__ base, int ld,
                                            int m, int k0, int lane, float s) {
  const float* p = base + (size_t)m * ld + k0 + ((lane >> 4) << 3);
  float t[16];
  ((float4*)t)[0] = *(const float4*)(p);
  ((float4*)t)[1] = *(const float4*)(p + 4);
  ((float4*)t)[2] = *(const float4*)(p + 16);
  ((float4*)t)[3] = *(const float4*)(p + 20);
  union { v16h v; _Float16 e[16]; } u;
#pragma unroll
  for (int i = 0; i < 16; ++i) u.e[i] = (_Float16)(t[i] * s);
  return u.v;
}

// overloads so the templated GEMM picks the right x_dst loader
__device__ __forceinline__ v16h load_a_any(const float* b, int ld, int m, int k0, int lane) {
  return load_a_f32s(b, ld, m, k0, lane, 1.0f);
}
__device__ __forceinline__ v16h load_a_any(const _Float16* b, int ld, int m, int k0, int lane) {
  return load_a_f16(b, ld, m, k0, lane);
}

// B (KxN) sourced from row-major W[N][K] (we need W^T): lane l holds column n = W row n.
//   lanes 0-15:  K = k0 + {0..7, 16..23}
//   lanes 16-31: K = k0 + {8..15, 24..31}
__device__ __forceinline__ v16h load_b_f16(const _Float16* __restrict__ W, int ld,
                                           int n, int k0, int lane) {
  const _Float16* p = W + (size_t)n * ld + k0 + ((lane >> 4) << 3);
  union { v16h v; v8h h[2]; } u;
  u.h[0] = *(const v8h*)(p);
  u.h[1] = *(const v8h*)(p + 16);
  return u.v;
}

// ---------- utility kernels ----------
__global__ void zero_f32_kernel(float* __restrict__ p, long long n) {
  long long i = (long long)blockIdx.x * blockDim.x + threadIdx.x;
  if (i < n) p[i] = 0.0f;
}

__global__ void f32_to_f16_kernel(const float* __restrict__ s, _Float16* __restrict__ d, int n) {
  int i = blockIdx.x * blockDim.x + threadIdx.x;
  if (i < n) d[i] = (_Float16)s[i];
}

// ---------- edge aggregation: one wave per edge, f32 atomics ----------
template <typename XT>
__global__ void sage_gather_kernel(const XT* __restrict__ xsrc, const int* __restrict__ row,
                                   const int* __restrict__ col, float* __restrict__ agg,
                                   float* __restrict__ cnt, int E, int K) {
  int lane = threadIdx.x & 31;
  long long w = ((long long)blockIdx.x * blockDim.x + threadIdx.x) >> 5;
  if (w >= E) return;
  int r = row[w];
  int c = col[w];
  const XT* s = xsrc + (long long)r * K;
  float* d = agg + (long long)c * K;
  for (int k = lane; k < K; k += 32) atomicAdd(d + k, (float)s[k]);
  if (lane == 0) atomicAdd(cnt + c, 1.0f);
}

// ---------- fused (mean-agg @ Wl^T + x_dst @ Wr^T) -> BN(eval) -> ReLU -> f16 ----------
// one 16x16 tile per wave; K in {128,256}; out is N x 256 (f16)
template <typename XT>
__global__ __launch_bounds__(256)
void sage_gemm_bn_relu(const float* __restrict__ agg, const float* __restrict__ cnt,
                       const XT* __restrict__ xdst,
                       const _Float16* __restrict__ Wl, const _Float16* __restrict__ Wr,
                       const float* __restrict__ g, const float* __restrict__ bb,
                       _Float16* __restrict__ out, int N, int K) {
  int lane = threadIdx.x & 31;
  int wave = blockIdx.x * 8 + (threadIdx.x >> 5);
  int Mt = (N + 15) >> 4;
  if (wave >= Mt * 16) return;                  // wave-uniform: EXEC stays all-ones
  int mt = wave >> 4;
  int nt = wave & 15;
  int m  = mt * 16 + (lane & 15);
  int mc = imind(m, N - 1);
  int n  = nt * 16 + (lane & 15);

  float s = 1.0f / fmaxf(cnt[mc], 1.0f);        // fold mean into A-load
  v8f acc = {};
  for (int k0 = 0; k0 < K; k0 += 32) {
    v16h a = load_a_f32s(agg, K, mc, k0, lane, s);
    v16h w = load_b_f16(Wl, K, n, k0, lane);
    acc = __builtin_amdgcn_wmma_f32_16x16x32_f16(false, a, false, w, (short)0, acc, false, false);
  }
  for (int k0 = 0; k0 < K; k0 += 32) {
    v16h a = load_a_any(xdst, K, mc, k0, lane);
    v16h w = load_b_f16(Wr, K, n, k0, lane);
    acc = __builtin_amdgcn_wmma_f32_16x16x32_f16(false, a, false, w, (short)0, acc, false, false);
  }

  float scale = BN_SCALE * g[n];
  float bias  = bb[n];
  int mb = mt * 16 + ((lane >> 4) << 3);        // C/D layout: M = r + 8*(lane/16)
#pragma unroll
  for (int r = 0; r < 8; ++r) {
    int mr = mb + r;
    if (mr < N) {
      float v = fmaxf(acc[r] * scale + bias, 0.0f);
      out[(size_t)mr * HIDC + n] = (_Float16)v;
    }
  }
}

// ---------- final: concat(h0,h1,h2)[:25000] @ Wlin^T + blin -> d_out (25000 x 47) ----------
__global__ __launch_bounds__(256)
void final_gemm_logits(const _Float16* __restrict__ h0, const _Float16* __restrict__ h1,
                       const _Float16* __restrict__ h2, const _Float16* __restrict__ Wlin,
                       const float* __restrict__ blin, float* __restrict__ z, int N) {
  int lane = threadIdx.x & 31;
  int wave = blockIdx.x * 8 + (threadIdx.x >> 5);
  int Mt = (N + 15) >> 4;
  if (wave >= Mt * 3) return;                   // 3 column tiles cover 48 >= 47
  int mt = wave / 3;
  int nt = wave % 3;
  int m  = mt * 16 + (lane & 15);
  int mc = imind(m, N - 1);
  int j  = nt * 16 + (lane & 15);
  int jc = imind(j, OUTC - 1);

  const _Float16* segs[3] = {h0, h1, h2};
  v8f acc = {};
  for (int k0 = 0; k0 < 3 * HIDC; k0 += 32) {
    const _Float16* seg = segs[k0 >> 8];        // 256 % 32 == 0: chunk never crosses segment
    int kloc = k0 & (HIDC - 1);
    v16h a = load_a_f16(seg, HIDC, mc, kloc, lane);
    v16h w = load_b_f16(Wlin, 3 * HIDC, jc, k0, lane);
    acc = __builtin_amdgcn_wmma_f32_16x16x32_f16(false, a, false, w, (short)0, acc, false, false);
  }

  int n = nt * 16 + (lane & 15);
  float bias = (n < OUTC) ? blin[n] : 0.0f;
  int mb = mt * 16 + ((lane >> 4) << 3);
#pragma unroll
  for (int r = 0; r < 8; ++r) {
    int mr = mb + r;
    if (mr < N && n < OUTC) z[(size_t)mr * OUTC + n] = acc[r] + bias;
  }
}

// ---------- in-place log-softmax over 47 classes, one thread per row ----------
__global__ void logsoftmax47_kernel(float* __restrict__ z, int N) {
  int i = blockIdx.x * blockDim.x + threadIdx.x;
  if (i >= N) return;
  float* p = z + (size_t)i * OUTC;
  float mx = p[0];
#pragma unroll
  for (int j = 1; j < OUTC; ++j) mx = fmaxf(mx, p[j]);
  float sum = 0.0f;
#pragma unroll
  for (int j = 0; j < OUTC; ++j) sum += expf(p[j] - mx);
  float l = logf(sum) + mx;
#pragma unroll
  for (int j = 0; j < OUTC; ++j) p[j] = p[j] - mx - (l - mx);
}

// ---------- host ----------
extern "C" void kernel_launch(void* const* d_in, const int* in_sizes, int n_in,
                              void* d_out, int out_size, void* d_ws, size_t ws_size,
                              hipStream_t stream) {
  const float* x    = (const float*)d_in[0];
  const int* row0   = (const int*)d_in[1];
  const int* col0   = (const int*)d_in[2];
  const int* row1   = (const int*)d_in[3];
  const int* col1   = (const int*)d_in[4];
  const int* row2   = (const int*)d_in[5];
  const int* col2   = (const int*)d_in[6];
  const float* Wl0  = (const float*)d_in[7];
  const float* Wr0  = (const float*)d_in[8];
  const float* g0   = (const float*)d_in[9];
  const float* b0   = (const float*)d_in[10];
  const float* Wl1  = (const float*)d_in[11];
  const float* Wr1  = (const float*)d_in[12];
  const float* g1   = (const float*)d_in[13];
  const float* b1   = (const float*)d_in[14];
  const float* Wl2  = (const float*)d_in[15];
  const float* Wr2  = (const float*)d_in[16];
  const float* g2   = (const float*)d_in[17];
  const float* b2   = (const float*)d_in[18];
  const float* Wlin = (const float*)d_in[19];
  const float* blin = (const float*)d_in[20];

  const int E0 = in_sizes[1], E1 = in_sizes[3], E2 = in_sizes[5];
  const int N0 = 100000, N1 = 50000, N2 = 25000;
  const int K0 = 128;

  // workspace carving (~142 MB total), 256B aligned
  char* p = (char*)d_ws;
  auto carve = [&](size_t bytes) -> char* {
    char* q = p;
    p += (bytes + 255) & ~(size_t)255;
    return q;
  };
  _Float16* Wl0h  = (_Float16*)carve((size_t)HIDC * K0 * 2);
  _Float16* Wr0h  = (_Float16*)carve((size_t)HIDC * K0 * 2);
  _Float16* Wl1h  = (_Float16*)carve((size_t)HIDC * HIDC * 2);
  _Float16* Wr1h  = (_Float16*)carve((size_t)HIDC * HIDC * 2);
  _Float16* Wl2h  = (_Float16*)carve((size_t)HIDC * HIDC * 2);
  _Float16* Wr2h  = (_Float16*)carve((size_t)HIDC * HIDC * 2);
  _Float16* Wlinh = (_Float16*)carve((size_t)OUTC * 3 * HIDC * 2);
  float*    cnt   = (float*)carve((size_t)N0 * 4);
  float*    agg   = (float*)carve((size_t)N0 * K0 * 4);      // == N1*HIDC*4, reused per layer
  _Float16* h0    = (_Float16*)carve((size_t)N0 * HIDC * 2);
  _Float16* h1    = (_Float16*)carve((size_t)N1 * HIDC * 2);
  _Float16* h2    = (_Float16*)carve((size_t)N2 * HIDC * 2);

  // weight conversions f32 -> f16
  f32_to_f16_kernel<<<(HIDC * K0 + 255) / 256, 256, 0, stream>>>(Wl0, Wl0h, HIDC * K0);
  f32_to_f16_kernel<<<(HIDC * K0 + 255) / 256, 256, 0, stream>>>(Wr0, Wr0h, HIDC * K0);
  f32_to_f16_kernel<<<(HIDC * HIDC + 255) / 256, 256, 0, stream>>>(Wl1, Wl1h, HIDC * HIDC);
  f32_to_f16_kernel<<<(HIDC * HIDC + 255) / 256, 256, 0, stream>>>(Wr1, Wr1h, HIDC * HIDC);
  f32_to_f16_kernel<<<(HIDC * HIDC + 255) / 256, 256, 0, stream>>>(Wl2, Wl2h, HIDC * HIDC);
  f32_to_f16_kernel<<<(HIDC * HIDC + 255) / 256, 256, 0, stream>>>(Wr2, Wr2h, HIDC * HIDC);
  f32_to_f16_kernel<<<(OUTC * 3 * HIDC + 255) / 256, 256, 0, stream>>>(Wlin, Wlinh, OUTC * 3 * HIDC);

  // ---- layer 0: x (f32, 200000x128) -> h0 (f16, 100000x256) ----
  {
    long long na = (long long)N0 * K0;
    zero_f32_kernel<<<(unsigned)((na + 255) / 256), 256, 0, stream>>>(agg, na);
    zero_f32_kernel<<<(N0 + 255) / 256, 256, 0, stream>>>(cnt, N0);
    sage_gather_kernel<float><<<(E0 + 7) / 8, 256, 0, stream>>>(x, row0, col0, agg, cnt, E0, K0);
    int tiles = ((N0 + 15) / 16) * 16;
    sage_gemm_bn_relu<float><<<(tiles + 7) / 8, 256, 0, stream>>>(
        agg, cnt, x, Wl0h, Wr0h, g0, b0, h0, N0, K0);
  }

  // ---- layer 1: h0 (f16) -> h1 (f16, 50000x256) ----
  {
    long long na = (long long)N1 * HIDC;
    zero_f32_kernel<<<(unsigned)((na + 255) / 256), 256, 0, stream>>>(agg, na);
    zero_f32_kernel<<<(N1 + 255) / 256, 256, 0, stream>>>(cnt, N1);
    sage_gather_kernel<_Float16><<<(E1 + 7) / 8, 256, 0, stream>>>(h0, row1, col1, agg, cnt, E1, HIDC);
    int tiles = ((N1 + 15) / 16) * 16;
    sage_gemm_bn_relu<_Float16><<<(tiles + 7) / 8, 256, 0, stream>>>(
        agg, cnt, h0, Wl1h, Wr1h, g1, b1, h1, N1, HIDC);
  }

  // ---- layer 2: h1 (f16) -> h2 (f16, 25000x256) ----
  {
    long long na = (long long)N2 * HIDC;
    zero_f32_kernel<<<(unsigned)((na + 255) / 256), 256, 0, stream>>>(agg, na);
    zero_f32_kernel<<<(N2 + 255) / 256, 256, 0, stream>>>(cnt, N2);
    sage_gather_kernel<_Float16><<<(E2 + 7) / 8, 256, 0, stream>>>(h1, row2, col2, agg, cnt, E2, HIDC);
    int tiles = ((N2 + 15) / 16) * 16;
    sage_gemm_bn_relu<_Float16><<<(tiles + 7) / 8, 256, 0, stream>>>(
        agg, cnt, h1, Wl2h, Wr2h, g2, b2, h2, N2, HIDC);
  }

  // ---- final: concat -> linear -> log_softmax ----
  {
    int tiles = ((N2 + 15) / 16) * 3;
    final_gemm_logits<<<(tiles + 7) / 8, 256, 0, stream>>>(h0, h1, h2, Wlinh, blin,
                                                           (float*)d_out, N2);
    logsoftmax47_kernel<<<(N2 + 255) / 256, 256, 0, stream>>>((float*)d_out, N2);
  }
}
